// GAT2NO_NORM_82575041232961
// MI455X (gfx1250) — matrix operand order repaired and checked by reading, not compiled, not run on hardware
//
#include <hip/hip_runtime.h>
#include <hip/hip_bf16.h>

// ---------------------------------------------------------------------------
// GAT forward for MI455X (gfx1250, wave32).
// Phases: fp32->bf16 convert, WMMA bf16 GEMM (h = x*W^T), per-node attention
// dots, edge segment-softmax via float atomics, L2-resident scatter-sum,
// head-mean + bias + relu.
// ---------------------------------------------------------------------------

#define N_NODES 50000
#define N_EDGES 800000
#define IN_CH   128
#define HC      256   // HEADS * OUT_CH
#define OUT_CH  64
#define HEADS   4
#define NEG_SLOPE 0.2f

static_assert(N_NODES % 16 == 0, "M tiling assumes 16 | N_NODES");

typedef __attribute__((ext_vector_type(16))) __bf16 bf16x16;
typedef __attribute__((ext_vector_type(8)))  __bf16 bf16x8;
typedef __attribute__((ext_vector_type(8)))  float  f32x8;

__device__ __forceinline__ unsigned short f2bf(float f) {
    unsigned int u = __float_as_uint(f);
    u += 0x7FFFu + ((u >> 16) & 1u);        // round-to-nearest-even
    return (unsigned short)(u >> 16);
}

__device__ __forceinline__ float leaky(float v) {
    return v > 0.0f ? v : NEG_SLOPE * v;
}

// float atomic max via signed/unsigned integer monotonicity trick
__device__ __forceinline__ void atomicMaxF32(float* addr, float val) {
    if (val >= 0.0f)
        atomicMax((int*)addr, __float_as_int(val));
    else
        atomicMin((unsigned int*)addr, __float_as_uint(val));
}

// ---------------------------------------------------------------- convert ---
__global__ void cvt_bf16_kernel(const float* __restrict__ src,
                                unsigned short* __restrict__ dst, int n) {
    int i = (blockIdx.x * 256 + threadIdx.x) * 4;
    if (i + 3 < n) {
        float4 v = *(const float4*)(src + i);
        ushort4 o;
        o.x = f2bf(v.x); o.y = f2bf(v.y); o.z = f2bf(v.z); o.w = f2bf(v.w);
        *(ushort4*)(dst + i) = o;
    }
}

// ------------------------------------------------------------- WMMA GEMM ---
// h[m, n] = sum_k x[m,k] * W[n,k];  M = N_NODES, N = HC = 256, K = 128.
// Block: 512 threads = 16 waves; block owns a 16-row M tile, wave w owns
// N tile w. A tile staged in LDS once per block. Grid covers M exactly, so
// the epilogue stores are unconditional (no exec-mask churn).
__global__ void __launch_bounds__(512)
gat_gemm_kernel(const unsigned short* __restrict__ xb,
                const unsigned short* __restrict__ wb,
                float* __restrict__ h) {
    __shared__ unsigned short As[16 * IN_CH];   // 4 KB

    const int tid   = threadIdx.x;
    const int mBase = blockIdx.x * 16;

    // Cooperative A-tile load: 512 threads x 8 bytes = 16x128 bf16.
    {
        int r  = tid >> 5;               // 0..15 row of tile
        int k4 = (tid & 31) * 4;         // 0..124
        *(uint2*)&As[r * IN_CH + k4] =
            *(const uint2*)&xb[(size_t)(mBase + r) * IN_CH + k4];
    }
    __syncthreads();

    const int wave  = tid >> 5;          // 0..15 == N tile
    const int lane  = tid & 31;
    const int col   = lane & 15;         // N / M row-in-tile index
    const int sel   = lane >> 4;         // half-K selector

    f32x8 c = {};

#pragma unroll
    for (int ks = 0; ks < 4; ++ks) {     // K = 4 * 32
        const int k0 = ks * 32;
        // A fragment: lanes 0-15 -> K = k0+0..15
        //             lanes16-31 -> K = k0+16..31
        bf16x8 a0 = *(const bf16x8*)&As[col * IN_CH + k0 + sel * 16];
        bf16x8 a1 = *(const bf16x8*)&As[col * IN_CH + k0 + sel * 16 + 8];
        bf16x16 a = __builtin_shufflevector(a0, a1,
                        0, 1, 2, 3, 4, 5, 6, 7, 8, 9, 10, 11, 12, 13, 14, 15);

        // B fragment: lane n holds column n (= W row n), 16 contiguous K.
        const unsigned short* wrow =
            wb + (size_t)(wave * 16 + col) * IN_CH + k0 + sel * 16;
        bf16x16 b = *(const bf16x16*)wrow;

        c = __builtin_amdgcn_wmma_f32_16x16x32_bf16(
                false, a, false, b, (short)0, c, false, false);
    }

    // D layout: VGPR r -> M = r (lanes 0-15) / r+8 (lanes 16-31), N = col.
    float* hp = h + (size_t)(mBase + sel * 8) * HC + wave * 16 + col;
#pragma unroll
    for (int r = 0; r < 8; ++r)
        hp[(size_t)r * HC] = c[r];
}

// ------------------------------------------- per-node attention + m init ---
// One wave per node: lane l handles channels [8l, 8l+8) of the 256-wide row
// (head = l>>3). Reduce within 8-lane groups, init m with self-loop score.
__global__ void attn_node_kernel(const float* __restrict__ h,
                                 const float* __restrict__ att_src,
                                 const float* __restrict__ att_dst,
                                 float* __restrict__ a_src,
                                 float* __restrict__ a_dst,
                                 float* __restrict__ m) {
    const int wave = threadIdx.x >> 5;
    const int lane = threadIdx.x & 31;
    const int n    = blockIdx.x * 8 + wave;
    if (n >= N_NODES) return;

    const int base = lane * 8;
    const float* hp = h + (size_t)n * HC + base;
    float4 h0 = *(const float4*)(hp);
    float4 h1 = *(const float4*)(hp + 4);
    float4 s0 = *(const float4*)(att_src + base);
    float4 s1 = *(const float4*)(att_src + base + 4);
    float4 d0 = *(const float4*)(att_dst + base);
    float4 d1 = *(const float4*)(att_dst + base + 4);

    float s = h0.x * s0.x + h0.y * s0.y + h0.z * s0.z + h0.w * s0.w
            + h1.x * s1.x + h1.y * s1.y + h1.z * s1.z + h1.w * s1.w;
    float d = h0.x * d0.x + h0.y * d0.y + h0.z * d0.z + h0.w * d0.w
            + h1.x * d1.x + h1.y * d1.y + h1.z * d1.z + h1.w * d1.w;

    // reduce across the 8 lanes of each head group
    s += __shfl_xor(s, 1, 32); d += __shfl_xor(d, 1, 32);
    s += __shfl_xor(s, 2, 32); d += __shfl_xor(d, 2, 32);
    s += __shfl_xor(s, 4, 32); d += __shfl_xor(d, 4, 32);

    if ((lane & 7) == 0) {
        int hd = lane >> 3;
        a_src[n * HEADS + hd] = s;
        a_dst[n * HEADS + hd] = d;
        m[n * HEADS + hd] = leaky(s + d);   // self-loop score seeds segment max
    }
}

// -------------------------------------------------------- edge atomic max ---
__global__ void edge_max_kernel(const int* __restrict__ ei,
                                const float* __restrict__ a_src,
                                const float* __restrict__ a_dst,
                                float* __restrict__ m) {
    int e = blockIdx.x * 256 + threadIdx.x;
    if (e >= N_EDGES) return;
    int src = ei[e];
    int dst = ei[N_EDGES + e];
    float4 as = *(const float4*)(a_src + src * HEADS);
    float4 ad = *(const float4*)(a_dst + dst * HEADS);
    atomicMaxF32(&m[dst * HEADS + 0], leaky(as.x + ad.x));
    atomicMaxF32(&m[dst * HEADS + 1], leaky(as.y + ad.y));
    atomicMaxF32(&m[dst * HEADS + 2], leaky(as.z + ad.z));
    atomicMaxF32(&m[dst * HEADS + 3], leaky(as.w + ad.w));
}

// ---------------------------------- init denom / accum with self-loop term ---
// Block = node, thread = channel c in [0,256). Coefficient vectors are
// block-uniform -> scalar (SMEM) loads.
__global__ void init_accum_kernel(const float* __restrict__ h,
                                  const float* __restrict__ a_src,
                                  const float* __restrict__ a_dst,
                                  const float* __restrict__ m,
                                  float* __restrict__ denom,
                                  float* __restrict__ accum) {
    const int n = blockIdx.x;
    const int c = threadIdx.x;
    const int hd = c >> 6;

    float4 as = *(const float4*)(a_src + n * HEADS);   // uniform -> s_load
    float4 ad = *(const float4*)(a_dst + n * HEADS);
    float4 mm = *(const float4*)(m + n * HEADS);
    float exv[HEADS];
    exv[0] = __expf(leaky(as.x + ad.x) - mm.x);
    exv[1] = __expf(leaky(as.y + ad.y) - mm.y);
    exv[2] = __expf(leaky(as.z + ad.z) - mm.z);
    exv[3] = __expf(leaky(as.w + ad.w) - mm.w);

    accum[(size_t)n * HC + c] = exv[hd] * h[(size_t)n * HC + c];
    if (c < HEADS)
        denom[n * HEADS + c] = exv[c];
}

// ------------------------------------------------ edge scatter (dominant) ---
// One wave per edge; e/src/dst are wave-uniform -> force into SGPRs with
// readfirstlane so edge metadata comes in via scalar loads, leaving VMEM to
// the h[src] reads and the f32 atomics. Lane covers channels {lane, lane+32,
// ...} so atomics are line-contiguous; accum (51 MB) + h (51 MB) stay resident
// in the 192 MB L2.
__global__ void __launch_bounds__(256)
edge_scatter_kernel(const int* __restrict__ ei,
                    const float* __restrict__ h,
                    const float* __restrict__ a_src,
                    const float* __restrict__ a_dst,
                    const float* __restrict__ m,
                    float* __restrict__ denom,
                    float* __restrict__ accum) {
    const int lane = threadIdx.x & 31;
    const int e =
        __builtin_amdgcn_readfirstlane(blockIdx.x * 8 + (threadIdx.x >> 5));
    if (e >= N_EDGES) return;

    const int src = __builtin_amdgcn_readfirstlane(ei[e]);
    const int dst = __builtin_amdgcn_readfirstlane(ei[N_EDGES + e]);

    float4 as = *(const float4*)(a_src + src * HEADS);  // uniform -> s_load
    float4 ad = *(const float4*)(a_dst + dst * HEADS);
    float4 mm = *(const float4*)(m + dst * HEADS);
    float ex[HEADS];
    ex[0] = __expf(leaky(as.x + ad.x) - mm.x);
    ex[1] = __expf(leaky(as.y + ad.y) - mm.y);
    ex[2] = __expf(leaky(as.z + ad.z) - mm.z);
    ex[3] = __expf(leaky(as.w + ad.w) - mm.w);

    if (lane == 0) {
#pragma unroll
        for (int hd = 0; hd < HEADS; ++hd)
            atomicAdd(&denom[dst * HEADS + hd], ex[hd]);
    }

    const float* hs = h + (size_t)src * HC;
    float* ac = accum + (size_t)dst * HC;
#pragma unroll
    for (int i = 0; i < 8; ++i) {
        int c = i * 32 + lane;          // head = i >> 1
        atomicAdd(&ac[c], ex[i >> 1] * hs[c]);
    }
}

// ------------------------------------------------------------- finalize -----
__global__ void finalize_kernel(const float* __restrict__ accum,
                                const float* __restrict__ denom,
                                const float* __restrict__ bias,
                                float* __restrict__ out) {
    int t = blockIdx.x * 256 + threadIdx.x;
    if (t >= N_NODES * OUT_CH) return;
    int n = t >> 6;
    int c = t & 63;
    float4 dn = *(const float4*)(denom + n * HEADS);
    const float* ap = accum + (size_t)n * HC + c;
    float acc = ap[0]          / dn.x
              + ap[OUT_CH]     / dn.y
              + ap[2 * OUT_CH] / dn.z
              + ap[3 * OUT_CH] / dn.w;
    float v = 0.25f * acc + bias[c];
    out[t] = v > 0.0f ? v : 0.0f;
}

// ---------------------------------------------------------------------------
extern "C" void kernel_launch(void* const* d_in, const int* in_sizes, int n_in,
                              void* d_out, int out_size, void* d_ws, size_t ws_size,
                              hipStream_t stream) {
    const float* x       = (const float*)d_in[0];
    const int*   ei      = (const int*)  d_in[1];
    const float* W       = (const float*)d_in[2];
    const float* att_src = (const float*)d_in[3];
    const float* att_dst = (const float*)d_in[4];
    const float* bias    = (const float*)d_in[5];
    float* out = (float*)d_out;

    // Workspace layout (256-byte aligned slabs).
    char* ws = (char*)d_ws;
    size_t off = 0;
    auto take = [&](size_t bytes) {
        char* p = ws + off;
        off += (bytes + 255) & ~(size_t)255;
        return p;
    };
    unsigned short* xb    = (unsigned short*)take((size_t)N_NODES * IN_CH * 2);
    unsigned short* wb    = (unsigned short*)take((size_t)HC * IN_CH * 2);
    float* h     = (float*)take((size_t)N_NODES * HC * 4);
    float* a_src = (float*)take((size_t)N_NODES * HEADS * 4);
    float* a_dst = (float*)take((size_t)N_NODES * HEADS * 4);
    float* m     = (float*)take((size_t)N_NODES * HEADS * 4);
    float* denom = (float*)take((size_t)N_NODES * HEADS * 4);
    float* accum = (float*)take((size_t)N_NODES * HC * 4);
    (void)ws_size; (void)n_in; (void)in_sizes; (void)out_size;

    // 1) fp32 -> bf16 converts
    {
        int nx = N_NODES * IN_CH;
        cvt_bf16_kernel<<<(nx / 4 + 255) / 256, 256, 0, stream>>>(x, xb, nx);
        int nw = HC * IN_CH;
        cvt_bf16_kernel<<<(nw / 4 + 255) / 256, 256, 0, stream>>>(W, wb, nw);
    }
    // 2) projection GEMM (WMMA bf16)
    gat_gemm_kernel<<<N_NODES / 16, 512, 0, stream>>>(xb, wb, h);
    // 3) per-node attention halves + self-loop max seed
    attn_node_kernel<<<(N_NODES + 7) / 8, 256, 0, stream>>>(
        h, att_src, att_dst, a_src, a_dst, m);
    // 4) segment max over edges
    edge_max_kernel<<<(N_EDGES + 255) / 256, 256, 0, stream>>>(ei, a_src, a_dst, m);
    // 5) init denom/accum with self-loop contribution
    init_accum_kernel<<<N_NODES, HC, 0, stream>>>(h, a_src, a_dst, m, denom, accum);
    // 6) weighted scatter-sum over edges (L2-resident atomics)
    edge_scatter_kernel<<<(N_EDGES + 7) / 8, 256, 0, stream>>>(
        ei, h, a_src, a_dst, m, denom, accum);
    // 7) head mean + bias + relu
    finalize_kernel<<<(N_NODES * OUT_CH + 255) / 256, 256, 0, stream>>>(
        accum, denom, bias, out);
}
